// GalaxyClassifierS4D_23038204576000
// MI455X (gfx1250) — matrix operand order repaired and checked by reading, not compiled, not run on hardware
//
#include <hip/hip_runtime.h>
#include <hip/hip_bf16.h>
#include <math.h>

#define B_   128
#define CIN  3
#define HW_  64
#define L_   4096
#define H_   64
#define N2_  32
#define Q_   64
#define NCHUNK (L_/Q_)
#define BT   32
#define LDB  72   // padded transposed-B row stride (halves): 144B = 16B-aligned, conflict-free

typedef _Float16 f16;
typedef _Float16 v16h __attribute__((ext_vector_type(16)));
typedef _Float16 v8h  __attribute__((ext_vector_type(8)));
typedef _Float16 v4h  __attribute__((ext_vector_type(4)));
typedef float    v8f  __attribute__((ext_vector_type(8)));
typedef float    v4f_ __attribute__((ext_vector_type(4)));

__device__ __forceinline__ float gelu_exact(float x){
  return 0.5f * x * (1.0f + erff(x * 0.70710678118654752f));
}
__device__ __forceinline__ float sigmoidf_(float x){ return 1.0f / (1.0f + expf(-x)); }

// ---- WMMA fragment helpers (contiguous k order shared by A and B, wave32) ----
// A: 16x32 f16, row-major [M][K]. Per lane: 16 contiguous halves (k = 16*hf + e).
__device__ __forceinline__ v16h frag_a(const f16* s, int m0, int lda, int kb, int lane){
  const int hf = lane >> 4, rl = lane & 15;
  const f16* p = s + (m0 + rl)*lda + kb + hf*16;
  v16h a;
#pragma unroll
  for(int e = 0; e < 16; ++e) a[e] = p[e];
  return a;
}
// B: 32x16 f16 from TRANSPOSED storage [N][K] (ldk = LDB). Per lane: 16 contiguous halves.
__device__ __forceinline__ v16h frag_b_t(const f16* s, int ldk, int kb, int n0, int lane){
  const int hf = lane >> 4, rl = lane & 15;
  const f16* p = s + (n0 + rl)*ldk + kb + hf*16;
  v16h b;
#pragma unroll
  for(int e = 0; e < 16; ++e) b[e] = p[e];
  return b;
}
__device__ __forceinline__ v8f wmma16(v16h a, v16h b, v8f c){
  return __builtin_amdgcn_wmma_f32_16x16x32_f16(false, a, false, b, (short)0, c, false, false);
}

// ---- CDNA5 async global->LDS (GVS mode), tracked by ASYNCcnt ----
__device__ __forceinline__ void async_load_b128(unsigned lds_addr, unsigned goff, const float* base){
  asm volatile("global_load_async_to_lds_b128 %0, %1, %2"
               :: "v"(lds_addr), "v"(goff), "s"(base) : "memory");
}
__device__ __forceinline__ void wait_async_le2(){
  asm volatile("s_wait_asynccnt 0x2" ::: "memory");
}
__device__ __forceinline__ void wait_async0(){
  asm volatile("s_wait_asynccnt 0x0" ::: "memory");
}

// ---- Kernel 0: Hilbert curve flat indices (order 6) ----
__global__ void k_hilbert(int* __restrict__ hil){
  int d = blockIdx.x * blockDim.x + threadIdx.x;
  if(d >= L_) return;
  int t = d, x = 0, y = 0;
  for(int s = 1; s < HW_; s <<= 1){
    int rx = 1 & (t >> 1);
    int ry = 1 & (t ^ rx);
    if(ry == 0){
      if(rx == 1){ x = s - 1 - x; y = s - 1 - y; }
      int tmp = x; x = y; y = tmp;
    }
    x += s * rx; y += s * ry;
    t >>= 2;
  }
  hil[d] = x * HW_ + y;
}

// ---- Kernel 1: gather + input projection; produces u0 in (B,H,L) f32 ----
__global__ void k_embed(const float* __restrict__ x, const float* __restrict__ Wup,
                        const float* __restrict__ bup, const int* __restrict__ hil,
                        float* __restrict__ u0){
  const int b = blockIdx.x >> 4;
  const int l = ((blockIdx.x & 15) << 8) + threadIdx.x;
  const int p = hil[l];
  const float x0 = x[((size_t)(b*CIN + 0))*L_ + p];
  const float x1 = x[((size_t)(b*CIN + 1))*L_ + p];
  const float x2 = x[((size_t)(b*CIN + 2))*L_ + p];
  const size_t base = ((size_t)b * H_) * L_ + l;
#pragma unroll 4
  for(int h = 0; h < H_; ++h){
    float v = x0*Wup[h*3+0] + x1*Wup[h*3+1] + x2*Wup[h*3+2] + bup[h];
    u0[base + (size_t)h * L_] = v;
  }
}

// ---- Kernel 2 (per layer): scan coefficient matrices ----
__global__ void k_coeff(const float* __restrict__ log_dt, const float* __restrict__ C_re,
                        const float* __restrict__ C_im, const float* __restrict__ log_A_real,
                        const float* __restrict__ A_imag,
                        f16* __restrict__ Tm, f16* __restrict__ M1, f16* __restrict__ Wu,
                        float* __restrict__ wQ){
  const int h = blockIdx.x;
  const int j = threadIdx.x;   // 0..63
  __shared__ float sK[Q_];
  const float dt = expf(log_dt[h]);
  float kacc = 0.f;
  for(int n = 0; n < N2_; ++n){
    const float Are = -expf(log_A_real[h*N2_ + n]);
    const float Aim = A_imag[h*N2_ + n];
    const float dre = Are*dt, dim = Aim*dt;
    const float ej   = expf(dre*(float)j);
    const float wjre = ej*cosf(dim*(float)j);
    const float wjim = ej*sinf(dim*(float)j);
    const float e1  = expf(dre);
    const float wre = e1*cosf(dim);
    const float wim = e1*sinf(dim);
    const float nre = wre - 1.f, nim = wim;
    const float den = Are*Are + Aim*Aim;
    const float qre = (nre*Are + nim*Aim)/den;
    const float qim = (nim*Are - nre*Aim)/den;
    const float cr = C_re[h*N2_ + n], ci = C_im[h*N2_ + n];
    const float ckre = cr*qre - ci*qim;
    const float ckim = cr*qim + ci*qre;
    const float cwre = ckre*wjre - ckim*wjim;   // Ck * w^j
    const float cwim = ckre*wjim + ckim*wjre;
    kacc += 2.f * cwre;
    const float c1re = cwre*wre - cwim*wim;     // Ck * w^{j+1}
    const float c1im = cwre*wim + cwim*wre;
    M1[(h*Q_ + j)*Q_ + n]       = (f16)( 2.f*c1re);
    M1[(h*Q_ + j)*Q_ + N2_ + n] = (f16)(-2.f*c1im);
    Wu[(h*Q_ + n)*Q_ + (Q_-1-j)]        = (f16)wjre;
    Wu[(h*Q_ + N2_ + n)*Q_ + (Q_-1-j)]  = (f16)wjim;
    if(j == 0){
      const float eQ = expf(dre*(float)Q_);
      wQ[(h*N2_ + n)*2 + 0] = eQ*cosf(dim*(float)Q_);
      wQ[(h*N2_ + n)*2 + 1] = eQ*sinf(dim*(float)Q_);
    }
  }
  sK[j] = kacc;
  __syncthreads();
  for(int i2 = 0; i2 < Q_; ++i2)
    Tm[(h*Q_ + j)*Q_ + i2] = (j >= i2) ? (f16)sK[j - i2] : (f16)0.f;
}

// ---- Kernel 3 (per layer): chunked SSM scan + causal conv via WMMA ----
// Double-buffered async global->LDS pipeline over 64 sequential chunks.
__global__ void __launch_bounds__(256) k_scan(const float* __restrict__ uin,
                       f16* __restrict__ ybuf,
                       const f16* __restrict__ Tmg, const f16* __restrict__ M1g,
                       const f16* __restrict__ Wug, const float* __restrict__ wQ,
                       const float* __restrict__ Dv){
  const int h  = blockIdx.x >> 2;
  const int b0 = (blockIdx.x & 3) * BT;
  const int t  = threadIdx.x;
  const int lane = t & 31, wid = t >> 5;
  const int tm = wid >> 1, tn = wid & 1;
  const int m0 = tm * 16, n0 = tn * 16;

  __shared__ __align__(16) f16   sT [Q_*Q_];
  __shared__ __align__(16) f16   sM1[Q_*Q_];
  __shared__ __align__(16) f16   sWu[Q_*Q_];
  __shared__ __align__(16) float sUt[2*BT*Q_];  // async target, double-buffered [b][l] f32
  __shared__ __align__(16) f16   sUh[BT*LDB];   // transposed B operand [b][k]
  __shared__ __align__(16) f16   sSh[BT*LDB];   // states f16 transposed [b][r]
  __shared__ __align__(16) f16   sYh[BT*LDB];   // output staging f16 [b][l]
  __shared__ float sS  [Q_*BT];                 // states f32 [row][b]
  __shared__ float sSin[Q_*BT];                 // Wu @ U

  for(int i = t; i < Q_*Q_/8; i += 256){
    *(v8h*)&sT [i*8] = *(const v8h*)&Tmg[h*Q_*Q_ + i*8];
    *(v8h*)&sM1[i*8] = *(const v8h*)&M1g[h*Q_*Q_ + i*8];
    *(v8h*)&sWu[i*8] = *(const v8h*)&Wug[h*Q_*Q_ + i*8];
  }
  for(int i = t; i < Q_*BT; i += 256) sS[i] = 0.f;
  for(int i = t; i < BT*LDB; i += 256) sSh[i] = (f16)0.f;
  const float Dh = Dv[h];
  __syncthreads();

  // hoist chunk-invariant A fragments into registers (6 x v16h)
  v16h aT[2], aM[2], aW[2];
#pragma unroll
  for(int kk = 0; kk < 2; ++kk){
    aT[kk] = frag_a(sT,  m0, Q_, kk*32, lane);
    aM[kk] = frag_a(sM1, m0, Q_, kk*32, lane);
    aW[kk] = frag_a(sWu, m0, Q_, kk*32, lane);
  }

  const float* ubase = uin + ((size_t)b0*H_ + h)*L_;
  const unsigned sU_lds = (unsigned)(uintptr_t)(void*)sUt;
  const int bl16 = t >> 4, seg16 = t & 15;        // 16B-segment decomposition

  // issue chunk 0 (2 x b128 per thread: 32b x 64l x 4B tile)
#pragma unroll
  for(int q = 0; q < 2; ++q){
    int bl = bl16 + q*16;
    unsigned goff  = (unsigned)((bl*H_*L_)*4 + seg16*16);
    unsigned laddr = sU_lds + (unsigned)(bl*256 + seg16*16);
    async_load_b128(laddr, goff, ubase);
  }

  for(int c = 0; c < NCHUNK; ++c){
    const int cur = c & 1;
    if(c + 1 < NCHUNK){
      // prefetch next chunk into the other buffer, then wait only for chunk c
#pragma unroll
      for(int q = 0; q < 2; ++q){
        int bl = bl16 + q*16;
        unsigned goff  = (unsigned)((bl*H_*L_ + (c+1)*Q_)*4 + seg16*16);
        unsigned laddr = sU_lds + (unsigned)((cur^1)*BT*Q_*4 + bl*256 + seg16*16);
        async_load_b128(laddr, goff, ubase);
      }
      wait_async_le2();
    } else {
      wait_async0();
    }
    __syncthreads();
    const float* sUcur = sUt + cur*BT*Q_;

    // f32 -> f16 transposed operand (vectorized: b128 load, b64 store)
    for(int q = 0; q < 2; ++q){
      int idx = q*256 + t;            // 512 x 4-element groups
      int bl = idx >> 4, g = idx & 15;
      v4f_ fv = *(const v4f_*)&sUcur[bl*Q_ + g*4];
      v4h  hv;
#pragma unroll
      for(int k = 0; k < 4; ++k) hv[k] = (f16)fv[k];
      *(v4h*)&sUh[bl*LDB + g*4] = hv;
    }
    __syncthreads();

    // Y = T@U + M1@S ; Sin = Wu@U
    v8f accY = {}; v8f accS = {};
#pragma unroll
    for(int kk = 0; kk < 2; ++kk){
      v16h bu = frag_b_t(sUh, LDB, kk*32, n0, lane);
      accY = wmma16(aT[kk], bu, accY);
      v16h bs = frag_b_t(sSh, LDB, kk*32, n0, lane);
      accY = wmma16(aM[kk], bs, accY);
      accS = wmma16(aW[kk], bu, accS);
    }
    {
      const int hf = lane >> 4, rl = lane & 15;
#pragma unroll
      for(int r = 0; r < 8; ++r){
        int row = m0 + r + 8*hf;     // C/D layout: M = r + 8*(lane>=16)
        int col = n0 + rl;
        sSin[row*BT + col] = accS[r];
        float yv = accY[r] + Dh * sUcur[col*Q_ + row];
        sYh[col*LDB + row] = (f16)gelu_exact(yv);   // inner gelu, staged transposed
      }
    }
    __syncthreads();

    // state update s <- w^Q * s + Sin  (each (n,b) owned by exactly one thread)
    for(int q = 0; q < 4; ++q){
      int idx = q*256 + t;
      int n = idx >> 5, bl = idx & 31;
      float sre = sS[n*BT + bl];
      float sim = sS[(N2_ + n)*BT + bl];
      float wr = wQ[(h*N2_ + n)*2 + 0];
      float wi = wQ[(h*N2_ + n)*2 + 1];
      float nr = wr*sre - wi*sim + sSin[n*BT + bl];
      float ni = wi*sre + wr*sim + sSin[(N2_ + n)*BT + bl];
      sS[n*BT + bl] = nr;  sS[(N2_ + n)*BT + bl] = ni;
      sSh[bl*LDB + n] = (f16)nr;  sSh[bl*LDB + N2_ + n] = (f16)ni;
    }
    // store gelu(y) f16 (coalesced b64 along l)
    for(int q = 0; q < 2; ++q){
      int idx = q*256 + t;            // 512 x 4-half groups
      int bl = idx >> 4, g = idx & 15;
      v4h hv = *(const v4h*)&sYh[bl*LDB + g*4];
      *(v4h*)&ybuf[((size_t)(b0 + bl)*H_ + h)*L_ + c*Q_ + g*4] = hv;
    }
    __syncthreads();
  }
}

// ---- Kernel 4 (per layer): z = Wout@y + bout ; GLU ; outer gelu ----
__global__ void __launch_bounds__(256) k_glu(const f16* __restrict__ ybuf,
                      const float* __restrict__ Wout, const float* __restrict__ bout,
                      float* __restrict__ uout){
  const int b  = blockIdx.x >> 6;
  const int l0 = (blockIdx.x & 63) * 64;
  const int t  = threadIdx.x;
  const int lane = t & 31, wid = t >> 5;
  __shared__ __align__(16) f16   sW [128*64];   // A: [o][h]
  __shared__ __align__(16) f16   sXt[64*LDB];   // B transposed: [l][h]
  __shared__ float sZ[128*64];
  for(int i = t; i < 128*16; i += 256){         // v4 convert of Wout
    v4f_ wv = *(const v4f_*)&Wout[i*4];
    v4h  hv;
#pragma unroll
    for(int k = 0; k < 4; ++k) hv[k] = (f16)wv[k];
    *(v4h*)&sW[i*4] = hv;
  }
  for(int i = t; i < 64*16; i += 256){          // v4h global loads, transposed scatter
    int hh = i >> 4, jj = i & 15;
    v4h vv = *(const v4h*)&ybuf[((size_t)b*H_ + hh)*L_ + l0 + jj*4];
#pragma unroll
    for(int k = 0; k < 4; ++k) sXt[(jj*4 + k)*LDB + hh] = vv[k];
  }
  __syncthreads();
  const int m0 = wid * 16;      // 8 waves cover o = 0..127
  v8f acc[4];
#pragma unroll
  for(int tn = 0; tn < 4; ++tn) acc[tn] = (v8f){};
#pragma unroll
  for(int ks = 0; ks < 64; ks += 32){
    v16h a = frag_a(sW, m0, 64, ks, lane);
#pragma unroll
    for(int tn = 0; tn < 4; ++tn){
      v16h bb = frag_b_t(sXt, LDB, ks, tn*16, lane);
      acc[tn] = wmma16(a, bb, acc[tn]);
    }
  }
  const int hf = lane >> 4, rl = lane & 15;
#pragma unroll
  for(int tn = 0; tn < 4; ++tn)
#pragma unroll
    for(int r = 0; r < 8; ++r)
      sZ[(m0 + r + 8*hf)*64 + tn*16 + rl] = acc[tn][r];
  __syncthreads();
  for(int q = 0; q < 16; ++q){
    int idx = q*256 + t;
    int hh = idx >> 6, j = idx & 63;
    float z1 = sZ[hh*64 + j]        + bout[hh];
    float z2 = sZ[(hh + 64)*64 + j] + bout[64 + hh];
    float g  = z1 * sigmoidf_(z2);
    uout[((size_t)b*H_ + hh)*L_ + l0 + j] = gelu_exact(g);
  }
}

// ---- Kernel 5: classifier head + softmax ----
__global__ void k_head(const float* __restrict__ u, const float* __restrict__ Wfc,
                       const float* __restrict__ bfc, float* __restrict__ out){
  const int b = threadIdx.x;
  if(b >= B_) return;
  float last[H_];
  for(int h = 0; h < H_; ++h) last[h] = u[((size_t)b*H_ + h)*L_ + (L_ - 1)];
  float lg[4];
  float m = -1e30f;
  for(int c = 0; c < 4; ++c){
    float s = bfc[c];
    for(int h = 0; h < H_; ++h) s += last[h] * Wfc[c*H_ + h];
    lg[c] = s; m = fmaxf(m, s);
  }
  float den = 0.f;
  for(int c = 0; c < 4; ++c){ lg[c] = expf(lg[c] - m); den += lg[c]; }
  for(int c = 0; c < 4; ++c) out[b*4 + c] = lg[c] / den;
}

extern "C" void kernel_launch(void* const* d_in, const int* in_sizes, int n_in,
                              void* d_out, int out_size, void* d_ws, size_t ws_size,
                              hipStream_t stream){
  (void)in_sizes; (void)n_in; (void)out_size; (void)ws_size;
  const float* x      = (const float*)d_in[0];
  const float* Wup    = (const float*)d_in[1];
  const float* bup    = (const float*)d_in[2];
  const float* log_dt = (const float*)d_in[3];
  const float* C_re   = (const float*)d_in[4];
  const float* C_im   = (const float*)d_in[5];
  const float* lAr    = (const float*)d_in[6];
  const float* A_im   = (const float*)d_in[7];
  const float* Dv     = (const float*)d_in[8];
  const float* Wout   = (const float*)d_in[9];
  const float* bout   = (const float*)d_in[10];
  const float* Wfc    = (const float*)d_in[11];
  const float* bfc    = (const float*)d_in[12];
  float* out = (float*)d_out;

  char* ws = (char*)d_ws;
  size_t off = 0;
  auto take = [&](size_t bytes) -> char* {
    off = (off + 255) & ~(size_t)255;
    char* p = ws + off;
    off += bytes;
    return p;
  };
  int*   hil = (int*)  take((size_t)L_ * sizeof(int));
  float* u0  = (float*)take((size_t)B_*H_*L_*4);
  float* u1  = (float*)take((size_t)B_*H_*L_*4);
  f16*   yb  = (f16*)  take((size_t)B_*H_*L_*2);
  f16*   Tm  = (f16*)  take((size_t)H_*Q_*Q_*2);
  f16*   M1  = (f16*)  take((size_t)H_*Q_*Q_*2);
  f16*   Wu  = (f16*)  take((size_t)H_*Q_*Q_*2);
  float* wQ  = (float*)take((size_t)H_*N2_*2*4);

  k_hilbert<<<dim3(L_/256), dim3(256), 0, stream>>>(hil);
  k_embed<<<dim3(B_*(L_/256)), dim3(256), 0, stream>>>(x, Wup, bup, hil, u0);

  float* bufs[4] = {u0, u1, u0, u1};   // layer i: in bufs[i] -> out bufs[i+1]
  for(int i = 0; i < 3; ++i){
    k_coeff<<<dim3(H_), dim3(Q_), 0, stream>>>(
        log_dt + i*H_, C_re + (size_t)i*H_*N2_, C_im + (size_t)i*H_*N2_,
        lAr + (size_t)i*H_*N2_, A_im + (size_t)i*H_*N2_, Tm, M1, Wu, wQ);
    k_scan<<<dim3(H_*(B_/BT)), dim3(256), 0, stream>>>(
        bufs[i], yb, Tm, M1, Wu, wQ, Dv + i*H_);
    k_glu<<<dim3(B_*(L_/64)), dim3(256), 0, stream>>>(
        yb, Wout + (size_t)i*2*H_*H_, bout + (size_t)i*2*H_, bufs[i+1]);
  }
  k_head<<<dim3(1), dim3(B_), 0, stream>>>(bufs[3], Wfc, bfc, out);
}
